// GraphReadout_55963423867393
// MI455X (gfx1250) — matrix operand order repaired and checked by reading, not compiled, not run on hardware
//
#include <hip/hip_runtime.h>
#include <math.h>

#define HDIM 128
#define CHUNK 256

typedef __attribute__((ext_vector_type(16))) __bf16 v16bf;
typedef __attribute__((ext_vector_type(8)))  float  v8f;

union FragB { unsigned u[8]; v16bf v; };

// ---- truncating hi/lo bf16 split, packed pair via v_perm_b32 ----
// hiP = {bf16_trunc(b), bf16_trunc(a)}   (a in [15:0], b in [31:16])
// loP = {bf16_trunc(b-hi_b), bf16_trunc(a-hi_a)}
__device__ __forceinline__ void splitPackT(float a, float b, unsigned& hiP, unsigned& loP) {
  unsigned ua = __float_as_uint(a), ub = __float_as_uint(b);
  float ra = a - __uint_as_float(ua & 0xFFFF0000u);
  float rb = b - __uint_as_float(ub & 0xFFFF0000u);
  hiP = __builtin_amdgcn_perm(ub, ua, 0x07060302u);
  loP = __builtin_amdgcn_perm(__float_as_uint(rb), __float_as_uint(ra), 0x07060302u);
}

// ---- order-preserving float<->uint encoding (for atomicMax on floats) ----
__device__ __forceinline__ unsigned encF(float f) {
  unsigned u = __float_as_uint(f);
  return (u & 0x80000000u) ? ~u : (u | 0x80000000u);
}
__device__ __forceinline__ float decF(unsigned e) {
  return (e & 0x80000000u) ? __uint_as_float(e ^ 0x80000000u) : __uint_as_float(~e);
}
#define ENC_NEG_INF 0x007FFFFFu   /* encF(-inf) */

// ====================== kernel 0: init workspace ======================
__global__ void init_ws(float* segSum, unsigned* segMax, float* attnNum,
                        unsigned* counts, float* denom, unsigned* gmaxEnc, int B) {
  int i = blockIdx.x * blockDim.x + threadIdx.x;
  int tot = B * HDIM;
  if (i < tot) { segSum[i] = 0.f; attnNum[i] = 0.f; segMax[i] = ENC_NEG_INF; }
  if (i < B)   { counts[i] = 0u; denom[i] = 0.f; gmaxEnc[i] = ENC_NEG_INF; }
}

// ====================== kernel 1: gate MLP via WMMA ======================
// Each wave owns a 16-node tile: h[16,128] = x[16,128] @ W1[128,128] as
// 8 N-tiles x 4 K-chunks x 3 split-terms of v_wmma_f32_16x16x32_bf16.
// W1 staged once per workgroup into LDS in WMMA-B fragment layout (hi/lo bf16).
// jt loop only 2x-unrolled to keep VGPRs < 256 (avoids s_set_vgpr_msb churn).
__global__ __launch_bounds__(256) void gate_kernel(
    const float* __restrict__ x, const float* __restrict__ W1,
    const float* __restrict__ b1, const float* __restrict__ W2,
    const float* __restrict__ b2, const int* __restrict__ bi,
    float* __restrict__ gate, unsigned* __restrict__ gmaxEnc,
    int N, int numTiles) {
  extern __shared__ unsigned smem[];   // whi[8192], wlo[8192], b1[128], w2[128]
  unsigned* whi = smem;
  unsigned* wlo = smem + 8192;
  float* sB1 = (float*)(smem + 16384);
  float* sW2 = sB1 + HDIM;

  // ---- stage W1 -> LDS, WMMA-B layout: [frag=jt*4+kt][u:0..7][lane:0..31] ----
  // B (32x16 bf16): lanes 0-15 hold K=0..15 (pairs in u0..7), lanes 16-31 hold K=16..31.
  for (int l = threadIdx.x; l < 8192; l += blockDim.x) {
    int lane = l & 31;
    int u    = (l >> 5) & 7;
    int frag = l >> 8;
    int kt = frag & 3, jt = frag >> 2;
    int kk = kt * 32 + ((lane >= 16) ? 16 : 0) + u * 2;
    int n  = jt * 16 + (lane & 15);
    float a = W1[(size_t)kk * HDIM + n];
    float c = W1[(size_t)(kk + 1) * HDIM + n];
    splitPackT(a, c, whi[l], wlo[l]);
  }
  if (threadIdx.x < HDIM)        sB1[threadIdx.x] = b1[threadIdx.x];
  else if (threadIdx.x < 2*HDIM) sW2[threadIdx.x - HDIM] = W2[threadIdx.x - HDIM];
  __syncthreads();

  const int lane = threadIdx.x & 31;
  const int wid  = threadIdx.x >> 5;
  const int m    = lane & 15;
  const int half = lane >> 4;
  const float b2v = b2[0];

  for (int tile = blockIdx.x * 8 + wid; tile < numTiles; tile += gridDim.x * 8) {
    size_t row = (size_t)tile * 16 + m;
    bool valid = row < (size_t)N;
    const float* xr = x + row * HDIM;

    // ---- build A fragments (16x32 bf16 per K-chunk), hi/lo split ----
    // A layout: lanes 0-15: u0..7 <- K 0..15 ; lanes 16-31: K 16..31.
    FragB Ahi[4], Alo[4];
#pragma unroll
    for (int kt = 0; kt < 4; ++kt) {
      int k0 = kt * 32 + half * 16;
      float f0[8], f1[8];
      if (valid) {
        float4 q0 = *(const float4*)(xr + k0);
        float4 q1 = *(const float4*)(xr + k0 + 4);
        float4 q2 = *(const float4*)(xr + k0 + 8);
        float4 q3 = *(const float4*)(xr + k0 + 12);
        f0[0]=q0.x; f0[1]=q0.y; f0[2]=q0.z; f0[3]=q0.w;
        f0[4]=q1.x; f0[5]=q1.y; f0[6]=q1.z; f0[7]=q1.w;
        f1[0]=q2.x; f1[1]=q2.y; f1[2]=q2.z; f1[3]=q2.w;
        f1[4]=q3.x; f1[5]=q3.y; f1[6]=q3.z; f1[7]=q3.w;
      } else {
#pragma unroll
        for (int i = 0; i < 8; ++i) { f0[i] = 0.f; f1[i] = 0.f; }
      }
#pragma unroll
      for (int p = 0; p < 4; ++p) splitPackT(f0[2*p], f0[2*p+1], Ahi[kt].u[p],   Alo[kt].u[p]);
#pragma unroll
      for (int p = 0; p < 4; ++p) splitPackT(f1[2*p], f1[2*p+1], Ahi[kt].u[4+p], Alo[kt].u[4+p]);
    }

    // ---- matmul + silu + W2 dot ----
    float g[8] = {0.f,0.f,0.f,0.f,0.f,0.f,0.f,0.f};
#pragma unroll 2
    for (int jt = 0; jt < 8; ++jt) {
      v8f c = {0.f,0.f,0.f,0.f,0.f,0.f,0.f,0.f};
#pragma unroll
      for (int kt = 0; kt < 4; ++kt) {
        FragB Bh, Bl;
        int base = ((jt * 4 + kt) << 8) + lane;
#pragma unroll
        for (int u = 0; u < 8; ++u) { Bh.u[u] = whi[base + (u << 5)]; Bl.u[u] = wlo[base + (u << 5)]; }
        c = __builtin_amdgcn_wmma_f32_16x16x32_bf16(false, Ahi[kt].v, false, Bh.v, (short)0, c, false, false);
        c = __builtin_amdgcn_wmma_f32_16x16x32_bf16(false, Ahi[kt].v, false, Bl.v, (short)0, c, false, false);
        c = __builtin_amdgcn_wmma_f32_16x16x32_bf16(false, Alo[kt].v, false, Bh.v, (short)0, c, false, false);
      }
      float b1j = sB1[jt * 16 + m];
      float w2j = sW2[jt * 16 + m];
#pragma unroll
      for (int r = 0; r < 8; ++r) {
        float h = c[r] + b1j;
        float s = h / (1.f + __expf(-h));   // silu
        g[r] += s * w2j;
      }
    }

    // ---- reduce across the 16-lane half-groups (lanes 0-15: nodes 0-7, 16-31: nodes 8-15) ----
#pragma unroll
    for (int mask = 1; mask <= 8; mask <<= 1) {
#pragma unroll
      for (int r = 0; r < 8; ++r) g[r] += __shfl_xor(g[r], mask, 32);
    }

    if (m == 0) {
      size_t nb = (size_t)tile * 16 + half * 8;
#pragma unroll
      for (int r = 0; r < 8; ++r) {
        size_t node = nb + r;
        if (node < (size_t)N) {
          float gt = g[r] + b2v;
          gate[node] = gt;
          atomicMax(&gmaxEnc[bi[node]], encF(gt));
        }
      }
    }
  }
}

// ====================== kernel 2: streaming segment pooling ======================
// 128 threads = one feature per thread; sequential scan of a sorted node chunk,
// register run-accumulation with flush-on-segment-boundary (atomics ~ O(B), not O(N*H)).
__global__ __launch_bounds__(128) void pool_kernel(
    const float* __restrict__ x, const int* __restrict__ bi,
    const float* __restrict__ gate, const unsigned* __restrict__ gmaxEnc,
    float* __restrict__ segSum, unsigned* __restrict__ segMaxEnc,
    float* __restrict__ attnNum, unsigned* __restrict__ counts,
    float* __restrict__ denom, int N) {
  const int f = threadIdx.x;
  int start = blockIdx.x * CHUNK;
  int end = start + CHUNK; if (end > N) end = N;
  if (start >= end) return;

  int cur = bi[start];
  float gmCur = decF(gmaxEnc[cur]);
  float sSum = 0.f, sAtt = 0.f, sMax = -INFINITY;
  float cDen = 0.f; unsigned cCnt = 0u;

  for (int n = start; n < end; ++n) {
    int seg = bi[n];
    if (seg != cur) {                                  // uniform branch (seg depends on n only)
      atomicAdd(&segSum[(size_t)cur * HDIM + f], sSum);
      atomicMax(&segMaxEnc[(size_t)cur * HDIM + f], encF(sMax));
      atomicAdd(&attnNum[(size_t)cur * HDIM + f], sAtt);
      if (f == 0) { atomicAdd(&counts[cur], cCnt); atomicAdd(&denom[cur], cDen); }
      sSum = 0.f; sAtt = 0.f; sMax = -INFINITY; cDen = 0.f; cCnt = 0u;
      cur = seg; gmCur = decF(gmaxEnc[cur]);
    }
    if (n + 8 < N) __builtin_prefetch(&x[(size_t)(n + 8) * HDIM + f], 0, 0);
    float xv = x[(size_t)n * HDIM + f];
    float e = __expf(gate[n] - gmCur);
    sSum += xv;
    sMax = fmaxf(sMax, xv);
    sAtt += e * xv;
    if (f == 0) { cCnt++; cDen += e; }
  }
  atomicAdd(&segSum[(size_t)cur * HDIM + f], sSum);
  atomicMax(&segMaxEnc[(size_t)cur * HDIM + f], encF(sMax));
  atomicAdd(&attnNum[(size_t)cur * HDIM + f], sAtt);
  if (f == 0) { atomicAdd(&counts[cur], cCnt); atomicAdd(&denom[cur], cDen); }
}

// ====================== kernel 3: finalize / concat ======================
__global__ void finalize_kernel(const float* __restrict__ segSum,
                                const unsigned* __restrict__ segMaxEnc,
                                const float* __restrict__ attnNum,
                                const unsigned* __restrict__ counts,
                                const float* __restrict__ denom,
                                float* __restrict__ out, int B) {
  int i = blockIdx.x * blockDim.x + threadIdx.x;
  if (i >= B * HDIM) return;
  int b = i >> 7, f = i & (HDIM - 1);
  size_t ob = (size_t)b * (3 * HDIM);
  float cnt = (float)counts[b];
  out[ob + f]            = segSum[i] / fmaxf(cnt, 1.f);
  out[ob + HDIM + f]     = decF(segMaxEnc[i]);
  float dn = denom[b];
  out[ob + 2 * HDIM + f] = (dn > 0.f) ? attnNum[i] / dn : 0.f;
}

// ====================== launch ======================
extern "C" void kernel_launch(void* const* d_in, const int* in_sizes, int n_in,
                              void* d_out, int out_size, void* d_ws, size_t ws_size,
                              hipStream_t stream) {
  (void)n_in; (void)ws_size;
  const float* x  = (const float*)d_in[0];
  const float* W1 = (const float*)d_in[1];
  const float* b1 = (const float*)d_in[2];
  const float* W2 = (const float*)d_in[3];
  const float* b2 = (const float*)d_in[4];
  const int*   bi = (const int*)d_in[5];
  const int N = in_sizes[0] / HDIM;
  const int B = out_size / (3 * HDIM);

  char* ws = (char*)d_ws;
  size_t off = 0;
  float*    segSum  = (float*)(ws + off);    off += (size_t)B * HDIM * 4;
  unsigned* segMax  = (unsigned*)(ws + off); off += (size_t)B * HDIM * 4;
  float*    attnNum = (float*)(ws + off);    off += (size_t)B * HDIM * 4;
  float*    gate    = (float*)(ws + off);    off += (size_t)N * 4;
  unsigned* gmaxEnc = (unsigned*)(ws + off); off += (size_t)B * 4;
  unsigned* counts  = (unsigned*)(ws + off); off += (size_t)B * 4;
  float*    denom   = (float*)(ws + off);    off += (size_t)B * 4;

  int initTot = B * HDIM;
  init_ws<<<(initTot + 255) / 256, 256, 0, stream>>>(segSum, segMax, attnNum,
                                                     counts, denom, gmaxEnc, B);

  int numTiles = (N + 15) / 16;
  int blocks1 = (numTiles + 7) / 8; if (blocks1 > 1024) blocks1 = 1024;
  gate_kernel<<<blocks1, 256, 66560, stream>>>(x, W1, b1, W2, b2, bi,
                                               gate, gmaxEnc, N, numTiles);

  int blocks2 = (N + CHUNK - 1) / CHUNK;
  pool_kernel<<<blocks2, 128, 0, stream>>>(x, bi, gate, gmaxEnc, segSum, segMax,
                                           attnNum, counts, denom, N);

  finalize_kernel<<<(B * HDIM + 255) / 256, 256, 0, stream>>>(segSum, segMax, attnNum,
                                                              counts, denom,
                                                              (float*)d_out, B);
}
